// RecursiveAutoEncoderTorch_21698174779531
// MI455X (gfx1250) — matrix operand order repaired and checked by reading, not compiled, not run on hardware
//
#include <hip/hip_runtime.h>
#include <hip/hip_bf16.h>
#include <stdint.h>

// Recursive autoencoder, restructured for MI455X (gfx1250):
//   Phase 1 (WMMA bf16): Z = leaves[1:] @ W_encR^T + b_enc           (8191x1024x1024)
//   Phase 2 (scan):      p_j = normalize(tanh(W_encL @ p_{j-1} + Z_j)) -- sequential,
//                        32 persistent blocks, fp32 GEMV, atomic grid barrier per step
//   Phase 3 (WMMA bf16): U = tanh(P @ W_d^T + b_d) for d1 and d2     (2 x 8191x1024x1024)
//   Phase 4:             per-row L2-normalize + squared error, deterministic reductions.
// GEMM waves are register-blocked 32x32 (2x2 WMMA tiles) for 2x L2 reuse.

#define L_LEN 8192
#define D_DIM 1024
#define SCAN_NBLK 32

typedef __bf16 bf16_t;
typedef bf16_t v16bf __attribute__((ext_vector_type(16)));
typedef float  v8f   __attribute__((ext_vector_type(8)));

union FragBF { v16bf v; uint4 q[2]; };

__device__ __forceinline__ unsigned short f2bf(float f) {
  unsigned u = __float_as_uint(f);
  unsigned r = u + 0x7FFFu + ((u >> 16) & 1u);   // round-to-nearest-even
  return (unsigned short)(r >> 16);
}

// ---------------- conversion kernels ----------------
__global__ void k_cvt_bf16(const float* __restrict__ src,
                           unsigned short* __restrict__ dst, int n) {
  int i = blockIdx.x * blockDim.x + threadIdx.x;
  if (i < n) dst[i] = f2bf(src[i]);
}

__global__ void k_cvt_bf16_strided(const float* __restrict__ src,
                                   unsigned short* __restrict__ dst,
                                   int rows, int cols, int srcLd, int srcOff) {
  int i = blockIdx.x * blockDim.x + threadIdx.x;
  if (i < rows * cols) {
    int r = i / cols, c = i - r * cols;
    dst[i] = f2bf(src[(size_t)r * srcLd + srcOff + c]);
  }
}

// ---------------- WMMA GEMM: Out[M,N] = A[M,K](bf16) * W[N,K]^T(bf16) + bias, opt tanh ----
// A-frag (16x32 bf16): lanes 0-15 row M=l, K = k0+0..7 then k0+16..23;
//                      lanes 16-31 row M=l-16, K = k0+8..15 then k0+24..31.
__device__ __forceinline__ v16bf load_a_frag(const unsigned short* __restrict__ A, int lda,
                                             int row, int k0, int lane) {
  const unsigned short* p = A + (size_t)row * lda + k0 + ((lane >> 4) << 3);
  FragBF f;
  f.q[0] = *reinterpret_cast<const uint4*>(p);
  f.q[1] = *reinterpret_cast<const uint4*>(p + 16);
  return f.v;
}
// B-frag (32x16 bf16): lane holds column n = l&15; K = k0 + 8*(l>>4) + 0..7 then k0 + 16 + 8*(l>>4) + 0..7.
__device__ __forceinline__ v16bf load_b_frag(const unsigned short* __restrict__ W, int ldw,
                                             int col, int k0, int lane) {
  const unsigned short* p = W + (size_t)col * ldw + k0 + ((lane >> 4) << 3);
  FragBF f;
  f.q[0] = *reinterpret_cast<const uint4*>(p);
  f.q[1] = *reinterpret_cast<const uint4*>(p + 16);
  return f.v;
}

__global__ void __launch_bounds__(256)
k_gemm_bf16(const unsigned short* __restrict__ A, int aRows, int aRowOff,
            const unsigned short* __restrict__ W,
            const float* __restrict__ bias,
            float* __restrict__ Out, int Mvalid, int N, int K, int applyTanh) {
  const int lane = threadIdx.x & 31;
  const int wave = threadIdx.x >> 5;
  const int tile = blockIdx.x * (blockDim.x >> 5) + wave;   // 32x32 macro-tile per wave
  const int nTilesN = N >> 5;
  const int tm = tile / nTilesN, tn = tile - tm * nTilesN;
  const int row0 = tm << 5, col0 = tn << 5;

  int aRow0 = row0 + (lane & 15) + aRowOff;
  int aRow1 = aRow0 + 16;
  if (aRow0 >= aRows) aRow0 = 0;            // clamp; stores are guarded below
  if (aRow1 >= aRows) aRow1 = 0;
  const int colB0 = col0 + (lane & 15);
  const int colB1 = colB0 + 16;

  v8f acc00 = {}, acc01 = {}, acc10 = {}, acc11 = {};
  for (int k0 = 0; k0 < K; k0 += 32) {
    v16bf a0 = load_a_frag(A, K, aRow0, k0, lane);
    v16bf a1 = load_a_frag(A, K, aRow1, k0, lane);
    v16bf b0 = load_b_frag(W, K, colB0, k0, lane);
    v16bf b1 = load_b_frag(W, K, colB1, k0, lane);
    acc00 = __builtin_amdgcn_wmma_f32_16x16x32_bf16(false, a0, false, b0, (short)0, acc00, false, false);
    acc01 = __builtin_amdgcn_wmma_f32_16x16x32_bf16(false, a0, false, b1, (short)0, acc01, false, false);
    acc10 = __builtin_amdgcn_wmma_f32_16x16x32_bf16(false, a1, false, b0, (short)0, acc10, false, false);
    acc11 = __builtin_amdgcn_wmma_f32_16x16x32_bf16(false, a1, false, b1, (short)0, acc11, false, false);
  }

  const float bv0 = bias[colB0];
  const float bv1 = bias[colB1];
  const int rb0 = row0 + ((lane >> 4) << 3);   // C/D: VGPR v -> row v (lanes 0-15) / v+8
  const int rb1 = rb0 + 16;

  auto store_tile = [&](const v8f& acc, int rbase, int col, float bv) {
#pragma unroll
    for (int v = 0; v < 8; ++v) {
      int r = rbase + v;
      if (r < Mvalid) {
        float x = acc[v] + bv;
        if (applyTanh) x = tanhf(x);
        Out[(size_t)r * N + col] = x;
      }
    }
  };
  store_tile(acc00, rb0, colB0, bv0);
  store_tile(acc01, rb0, colB1, bv1);
  store_tile(acc10, rb1, colB0, bv0);
  store_tile(acc11, rb1, colB1, bv1);
}

// ---------------- sequential scan ----------------
__global__ void k_init_sync(unsigned int* sync) {
  if (threadIdx.x < 2) sync[threadIdx.x] = 0u;
}

__global__ void __launch_bounds__(256)
k_scan(const float* __restrict__ emb, const float* __restrict__ Wenc,
       const float* __restrict__ Z, float* __restrict__ U,
       float* __restrict__ nrmPart, unsigned int* __restrict__ sync) {
  __shared__ float c1s[D_DIM];
  __shared__ float wpart[8];
  __shared__ float sInvSh;
  const int tid = threadIdx.x;
  const int lane = tid & 31;
  const int wave = tid >> 5;
  const int rowBase = blockIdx.x * (D_DIM / SCAN_NBLK);     // 32 rows per block
  const int n0 = rowBase + wave * 4;                        // 4 rows per wave
  const float* __restrict__ w0 = Wenc + (size_t)n0 * (2 * D_DIM);
  const int steps = L_LEN - 1;
  for (int j = 0; j < steps; ++j) {
    // build c1 = p_{j-1} (or leaf 0) in LDS
    if (j == 0) {
      for (int k = tid; k < D_DIM; k += 256) c1s[k] = emb[k];
    } else {
      if (tid == 0) {
        const float* np = nrmPart + (size_t)(j - 1) * SCAN_NBLK;
        float s = 0.f;
#pragma unroll
        for (int b = 0; b < SCAN_NBLK; ++b) s += np[b];     // fixed order: deterministic
        sInvSh = rsqrtf(s);
      }
      __syncthreads();
      const float sc = sInvSh;
      const float* up = U + (size_t)(j - 1) * D_DIM;
      for (int k = tid; k < D_DIM; k += 256) c1s[k] = up[k] * sc;
    }
    __syncthreads();
    // 4-row GEMV slice per wave, fp32 weights streamed from L2
    float a0 = 0.f, a1 = 0.f, a2 = 0.f, a3 = 0.f;
#pragma unroll 4
    for (int i = 0; i < 32; ++i) {
      const int k = lane + (i << 5);
      const float cv = c1s[k];
      a0 = fmaf(w0[k], cv, a0);
      a1 = fmaf(w0[2048 + k], cv, a1);
      a2 = fmaf(w0[4096 + k], cv, a2);
      a3 = fmaf(w0[6144 + k], cv, a3);
    }
#pragma unroll
    for (int off = 16; off > 0; off >>= 1) {
      a0 += __shfl_down(a0, off, 32);
      a1 += __shfl_down(a1, off, 32);
      a2 += __shfl_down(a2, off, 32);
      a3 += __shfl_down(a3, off, 32);
    }
    if (lane == 0) {
      const float* zr = Z + (size_t)j * D_DIM + n0;
      float* ur = U + (size_t)j * D_DIM + n0;
      float x0 = tanhf(a0 + zr[0]);
      float x1 = tanhf(a1 + zr[1]);
      float x2 = tanhf(a2 + zr[2]);
      float x3 = tanhf(a3 + zr[3]);
      ur[0] = x0; ur[1] = x1; ur[2] = x2; ur[3] = x3;
      wpart[wave] = x0 * x0 + x1 * x1 + x2 * x2 + x3 * x3;
    }
    __threadfence();                 // every thread: make its global stores visible
    __syncthreads();
    if (tid == 0) {
      float s = 0.f;
#pragma unroll
      for (int w = 0; w < 8; ++w) s += wpart[w];
      nrmPart[(size_t)j * SCAN_NBLK + blockIdx.x] = s;
      __threadfence();
      unsigned arrived = atomicAdd(&sync[0], 1u) + 1u;      // cumulative arrivals
      if (arrived == (unsigned)(j + 1) * SCAN_NBLK) atomicAdd(&sync[1], 1u);
      while (atomicAdd(&sync[1], 0u) < (unsigned)(j + 1)) __builtin_amdgcn_s_sleep(1);
      __threadfence();
    }
    __syncthreads();
  }
}

// ---------------- normalize: Pbf = bf16(U * rsqrt(norm)) ----------------
__global__ void k_normalize(const float* __restrict__ U,
                            const float* __restrict__ nrmPart,
                            unsigned short* __restrict__ Pbf) {
  const int j = blockIdx.x;
  __shared__ float sInvSh;
  if (threadIdx.x == 0) {
    const float* np = nrmPart + (size_t)j * SCAN_NBLK;
    float s = 0.f;
#pragma unroll
    for (int b = 0; b < SCAN_NBLK; ++b) s += np[b];
    sInvSh = rsqrtf(s);
  }
  __syncthreads();
  const float sc = sInvSh;
  const float* up = U + (size_t)j * D_DIM;
  unsigned short* pp = Pbf + (size_t)j * D_DIM;
  for (int k = threadIdx.x; k < D_DIM; k += blockDim.x) pp[k] = f2bf(up[k] * sc);
}

// ---------------- per-row error: 0.5 * || normalize(Y_j) - target_j ||^2 ----------------
__global__ void __launch_bounds__(256)
k_err(const float* __restrict__ Y, const float* __restrict__ U,
      const float* __restrict__ nrmPart, const float* __restrict__ emb,
      float* __restrict__ errArr, int mode) {         // mode 1: c1 targets (=), 2: c2 (+=)
  const int j = blockIdx.x;
  const int tid = threadIdx.x, lane = tid & 31, wave = tid >> 5;
  __shared__ float red[8];
  __shared__ float sySh, stSh;
  const float* y = Y + (size_t)j * D_DIM;
  float a = 0.f;
  for (int k = tid; k < D_DIM; k += 256) { float v = y[k]; a = fmaf(v, v, a); }
#pragma unroll
  for (int off = 16; off > 0; off >>= 1) a += __shfl_down(a, off, 32);
  if (lane == 0) red[wave] = a;
  __syncthreads();
  if (tid == 0) {
    float s = 0.f;
    for (int w = 0; w < 8; ++w) s += red[w];
    sySh = rsqrtf(s);
    float st = 1.f;
    if (mode == 1 && j > 0) {
      const float* np = nrmPart + (size_t)(j - 1) * SCAN_NBLK;
      float t = 0.f;
      for (int b = 0; b < SCAN_NBLK; ++b) t += np[b];
      st = rsqrtf(t);
    }
    stSh = st;
  }
  __syncthreads();
  const float sy = sySh, ts = stSh;
  const float* T = (mode == 1) ? ((j == 0) ? emb : (U + (size_t)(j - 1) * D_DIM))
                               : (emb + (size_t)(j + 1) * D_DIM);
  float e = 0.f;
  for (int k = tid; k < D_DIM; k += 256) {
    float d = y[k] * sy - T[k] * ts;
    e = fmaf(d, d, e);
  }
#pragma unroll
  for (int off = 16; off > 0; off >>= 1) e += __shfl_down(e, off, 32);
  if (lane == 0) red[wave] = e;
  __syncthreads();
  if (tid == 0) {
    float s = 0.f;
    for (int w = 0; w < 8; ++w) s += red[w];
    s *= 0.5f;
    if (mode == 1) errArr[j] = s; else errArr[j] += s;
  }
}

__global__ void k_final(const float* __restrict__ errArr, int n, float* __restrict__ out) {
  __shared__ float red[8];
  const int tid = threadIdx.x, lane = tid & 31, wave = tid >> 5;
  float a = 0.f;
  for (int k = tid; k < n; k += 256) a += errArr[k];   // fixed order per thread
#pragma unroll
  for (int off = 16; off > 0; off >>= 1) a += __shfl_down(a, off, 32);
  if (lane == 0) red[wave] = a;
  __syncthreads();
  if (tid == 0) {
    float s = 0.f;
    for (int w = 0; w < 8; ++w) s += red[w];
    out[0] = s;
  }
}

// ---------------- host ----------------
extern "C" void kernel_launch(void* const* d_in, const int* in_sizes, int n_in,
                              void* d_out, int out_size, void* d_ws, size_t ws_size,
                              hipStream_t stream) {
  const float* emb   = (const float*)d_in[0];
  const float* W_enc = (const float*)d_in[1];
  const float* b_enc = (const float*)d_in[2];
  const float* W_d1  = (const float*)d_in[3];
  const float* b_d1  = (const float*)d_in[4];
  const float* W_d2  = (const float*)d_in[5];
  const float* b_d2  = (const float*)d_in[6];
  // d_in[7] (sKids) ignored: the tree is a fixed left chain.
  float* out = (float*)d_out;

  uintptr_t base = (uintptr_t)d_ws;
  auto take = [&](size_t bytes) {
    uintptr_t r = base; base += (bytes + 255) & ~(size_t)255; return r;
  };
  unsigned short* leavesBF = (unsigned short*)take((size_t)L_LEN * D_DIM * 2);
  unsigned short* WencRbf  = (unsigned short*)take((size_t)D_DIM * D_DIM * 2);
  unsigned short* Wd1bf    = (unsigned short*)take((size_t)D_DIM * D_DIM * 2);
  unsigned short* Wd2bf    = (unsigned short*)take((size_t)D_DIM * D_DIM * 2);
  float* Z   = (float*)take((size_t)L_LEN * D_DIM * 4);   // reused as decoder output
  float* U   = (float*)take((size_t)L_LEN * D_DIM * 4);   // un-normalized tanh outputs
  unsigned short* Pbf = (unsigned short*)take((size_t)L_LEN * D_DIM * 2);
  float* nrmPart = (float*)take((size_t)L_LEN * SCAN_NBLK * 4);
  float* errArr  = (float*)take((size_t)L_LEN * 4);
  unsigned int* syncBuf = (unsigned int*)take(256);

  // Phase 0: bf16 conversions
  k_cvt_bf16<<<(L_LEN * D_DIM) / 256, 256, 0, stream>>>(emb, leavesBF, L_LEN * D_DIM);
  k_cvt_bf16_strided<<<(D_DIM * D_DIM) / 256, 256, 0, stream>>>(
      W_enc, WencRbf, D_DIM, D_DIM, 2 * D_DIM, D_DIM);      // right half of W_enc
  k_cvt_bf16<<<(D_DIM * D_DIM) / 256, 256, 0, stream>>>(W_d1, Wd1bf, D_DIM * D_DIM);
  k_cvt_bf16<<<(D_DIM * D_DIM) / 256, 256, 0, stream>>>(W_d2, Wd2bf, D_DIM * D_DIM);

  const int tiles = (L_LEN / 32) * (D_DIM / 32);            // 8192 macro-tiles, 8 waves/block
  // Phase 1: Z = leaves[1:] @ W_encR^T + b_enc   (no tanh)
  k_gemm_bf16<<<tiles / 8, 256, 0, stream>>>(leavesBF, L_LEN, 1, WencRbf, b_enc,
                                             Z, L_LEN - 1, D_DIM, D_DIM, 0);
  // Phase 2: sequential scan
  k_init_sync<<<1, 64, 0, stream>>>(syncBuf);
  k_scan<<<SCAN_NBLK, 256, 0, stream>>>(emb, W_enc, Z, U, nrmPart, syncBuf);
  k_normalize<<<L_LEN - 1, 256, 0, stream>>>(U, nrmPart, Pbf);
  // Phase 3a: decoder 1 + error vs c1
  k_gemm_bf16<<<tiles / 8, 256, 0, stream>>>(Pbf, L_LEN - 1, 0, Wd1bf, b_d1,
                                             Z, L_LEN - 1, D_DIM, D_DIM, 1);
  k_err<<<L_LEN - 1, 256, 0, stream>>>(Z, U, nrmPart, emb, errArr, 1);
  // Phase 3b: decoder 2 + error vs c2
  k_gemm_bf16<<<tiles / 8, 256, 0, stream>>>(Pbf, L_LEN - 1, 0, Wd2bf, b_d2,
                                             Z, L_LEN - 1, D_DIM, D_DIM, 1);
  k_err<<<L_LEN - 1, 256, 0, stream>>>(Z, U, nrmPart, emb, errArr, 2);
  // Phase 4: deterministic final sum
  k_final<<<1, 256, 0, stream>>>(errArr, L_LEN - 1, out);

  (void)in_sizes; (void)n_in; (void)out_size; (void)ws_size;
}
